// RoIHead_79929341378761
// MI455X (gfx1250) — compile-verified, matrix operand order checked by reading, not run on hardware
//
#include <hip/hip_runtime.h>

// Toggle: stage GEMM tiles with GLOBAL_LOAD_ASYNC_TO_LDS_B128 (CDNA5 async
// copy, ASYNCcnt-tracked) instead of global->VGPR->LDS.
#define USE_ASYNC_TDM 1

// ---------------------------------------------------------------------------
// Problem constants (from reference setup_inputs)
// ---------------------------------------------------------------------------
#define OUTP   7
#define RNUM   512
#define CNUM   256
#define HNUM   50
#define WNUM   50
#define FC     1024
#define INFEAT (CNUM * OUTP * OUTP)   // 12544
#define NLOC   84
#define NSCORE 21

// GEMM tiling
#define BK    32
#define LDP   40            // padded LDS row stride in ushorts (32 data + 8 pad)
#define LDPW  20            // row stride in 32-bit words
#define BM    64            // fast-path block tile M
#define BN    128           // fast-path block tile N
#define BMF   64            // fallback block tile M
#define BNF   64            // fallback block tile N

// ---------------------------------------------------------------------------
// Trivially-constructible ext vectors (HIP structs have user ctors -> no unions)
// ---------------------------------------------------------------------------
typedef __attribute__((ext_vector_type(16))) __bf16       v16bf;
typedef __attribute__((ext_vector_type(8)))  float        v8f;
typedef __attribute__((ext_vector_type(4)))  float        v4f;
typedef __attribute__((ext_vector_type(4)))  unsigned int v4u;
typedef __attribute__((ext_vector_type(2)))  unsigned int v2u;

union FragBF {                // one 16x32 bf16 WMMA operand: 32B = 8 VGPRs
    v16bf v;
    v4u   q[2];
};

// Round-to-nearest-even f32 -> bf16 split: f ~= hi + lo (bf16 each)
__device__ __forceinline__ void bf16_split(float f, unsigned int& hi, unsigned int& lo) {
    unsigned int u = __float_as_uint(f);
    unsigned int h = (u + 0x7FFFu + ((u >> 16) & 1u)) >> 16;
    float fh = __uint_as_float(h << 16);
    float r  = f - fh;
    unsigned int v = __float_as_uint(r);
    unsigned int l = (v + 0x7FFFu + ((v >> 16) & 1u)) >> 16;
    hi = h;
    lo = l;
}

__device__ __forceinline__ void cvt_pack(v4f v, v2u& ph, v2u& pl) {
    unsigned int h0, l0, h1, l1, h2, l2, h3, l3;
    bf16_split(v.x, h0, l0); bf16_split(v.y, h1, l1);
    bf16_split(v.z, h2, l2); bf16_split(v.w, h3, l3);
    ph.x = h0 | (h1 << 16); ph.y = h2 | (h3 << 16);
    pl.x = l0 | (l1 << 16); pl.y = l2 | (l3 << 16);
}

#if USE_ASYNC_TDM
// 16B/lane async copy, global memory -> LDS, tracked with ASYNCcnt.
// Generic pointers into __shared__ carry the LDS byte offset in addr[31:0]
// (flat aperture rule), which is exactly what the VDST LDS-address VGPR wants.
__device__ __forceinline__ void async_cp16(void* lds_ptr, const void* gaddr) {
    const unsigned lds_off = (unsigned)(unsigned long long)lds_ptr;
    asm volatile("global_load_async_to_lds_b128 %0, %1, off"
                 :: "v"(lds_off), "v"(gaddr) : "memory");
}
__device__ __forceinline__ void wait_async0() {
    asm volatile("s_wait_asynccnt 0x0" ::: "memory");
}
#endif

// ---------------------------------------------------------------------------
// Stage 1: RoI max pool. Emits f32 (fallback) and/or pre-split bf16 hi/lo.
// ---------------------------------------------------------------------------
__global__ __launch_bounds__(256)
void roi_pool_kernel(const float* __restrict__ x,      // [B,C,H,W]
                     const float* __restrict__ rois,   // [R,4] (y1,x1,y2,x2)
                     const int*   __restrict__ ridx,   // [R]
                     float*          __restrict__ flatF,   // [R,INFEAT] or null
                     unsigned short* __restrict__ flatHi,  // [R,INFEAT] or null
                     unsigned short* __restrict__ flatLo)  // [R,INFEAT] or null
{
    const int r = blockIdx.x;
    // xyxy = rois[:, [1,0,3,2]]
    const float y1f = rois[r * 4 + 0];
    const float x1f = rois[r * 4 + 1];
    const float y2f = rois[r * 4 + 2];
    const float x2f = rois[r * 4 + 3];
    const int x1 = (int)rintf(x1f * 0.0625f);   // rintf == round-half-even == jnp.round
    const int y1 = (int)rintf(y1f * 0.0625f);
    const int x2 = (int)rintf(x2f * 0.0625f);
    const int y2 = (int)rintf(y2f * 0.0625f);
    const float roi_w = (float)((x2 - x1) > 1 ? (x2 - x1) : 1);
    const float roi_h = (float)((y2 - y1) > 1 ? (y2 - y1) : 1);
    const float bw = roi_w * (1.0f / (float)OUTP);
    const float bh = roi_h * (1.0f / (float)OUTP);
    const int b = ridx[r];
    const float* __restrict__ feat = x + (size_t)b * CNUM * HNUM * WNUM;

    for (int o = threadIdx.x; o < INFEAT; o += 256) {
        const int c   = o / (OUTP * OUTP);
        const int rem = o - c * (OUTP * OUTP);
        const int ph  = rem / OUTP;
        const int pw  = rem - ph * OUTP;

        int hs = (int)floorf((float)ph * bh) + y1;
        int he = (int)ceilf(((float)ph + 1.0f) * bh) + y1;
        int ws = (int)floorf((float)pw * bw) + x1;
        int we = (int)ceilf(((float)pw + 1.0f) * bw) + x1;
        hs = hs < 0 ? 0 : (hs > HNUM ? HNUM : hs);
        he = he < 0 ? 0 : (he > HNUM ? HNUM : he);
        ws = ws < 0 ? 0 : (ws > WNUM ? WNUM : ws);
        we = we < 0 ? 0 : (we > WNUM ? WNUM : we);

        float v = 0.0f;
        if (he > hs && we > ws) {
            float m = -__builtin_inff();
            const float* __restrict__ fc = feat + (size_t)c * HNUM * WNUM;
            for (int h = hs; h < he; ++h) {
                const float* rowp = fc + h * WNUM;
                for (int w = ws; w < we; ++w) m = fmaxf(m, rowp[w]);
            }
            v = m;
        }
        const size_t idx = (size_t)r * INFEAT + o;
        if (flatF) flatF[idx] = v;
        if (flatHi) {
            unsigned int h, l;
            bf16_split(v, h, l);
            flatHi[idx] = (unsigned short)h;
            flatLo[idx] = (unsigned short)l;
        }
    }
}

// ---------------------------------------------------------------------------
// f32 -> bf16 hi/lo split of a flat array (weights), vectorized x4.
// ---------------------------------------------------------------------------
__global__ __launch_bounds__(256)
void split_kernel(const float* __restrict__ in,
                  unsigned short* __restrict__ hi,
                  unsigned short* __restrict__ lo, int n4)
{
    const int i = blockIdx.x * 256 + threadIdx.x;
    if (i < n4) {
        const v4f v = ((const v4f*)in)[i];
        v2u ph, pl;
        cvt_pack(v, ph, pl);
        ((v2u*)hi)[i] = ph;
        ((v2u*)lo)[i] = pl;
    }
}

// ---------------------------------------------------------------------------
// Fast path GEMM: pre-split bf16x3, software pipelined, ping-pong LDS.
//   C = act(A[M,K] @ B[N,K]^T + bias)  with A,B given as bf16 hi/lo pairs.
// 128 threads (4 waves), 64x128 tile, BK=32. Each wave owns 32(M) x 64(N):
// 24 wmma per K-step against 24 ds_load_b128. Tile staging uses the CDNA5
// async global->LDS path (ASYNCcnt) so the next tile lands in LDS while the
// WMMA block runs; one s_wait_asynccnt + barrier per K-step.
// Requires M%64==0, N%128==0, K%32==0 (true for all calls here).
// ---------------------------------------------------------------------------
__global__ __launch_bounds__(128)
void gemm_bf16x3_pre(const unsigned short* __restrict__ Ahi,
                     const unsigned short* __restrict__ Alo,   // [M,K]
                     const unsigned short* __restrict__ Bhi,
                     const unsigned short* __restrict__ Blo,   // [N,K]
                     const float* __restrict__ bias,           // [N]
                     float*          __restrict__ Cf,          // [M,N] or null
                     unsigned short* __restrict__ Chi,         // [M,N] or null
                     unsigned short* __restrict__ Clo,         // [M,N] or null
                     int M, int N, int K, int relu)
{
    __shared__ __attribute__((aligned(16))) unsigned short sAhi[2][BM * LDP];
    __shared__ __attribute__((aligned(16))) unsigned short sAlo[2][BM * LDP];
    __shared__ __attribute__((aligned(16))) unsigned short sBhi[2][BN * LDP];
    __shared__ __attribute__((aligned(16))) unsigned short sBlo[2][BN * LDP];

    const int t    = threadIdx.x;
    const int wave = t >> 5;
    const int lane = t & 31;
    const int l16  = lane & 15;
    const int sel  = lane >> 4;
    const int wm   = wave >> 1;    // 0..1 : 32-row stripe
    const int wn   = wave & 1;     // 0..1 : 64-col stripe

    const int m0 = blockIdx.y * BM;
    const int n0 = blockIdx.x * BN;

    const unsigned short* __restrict__ AhiB = Ahi + (size_t)m0 * K;
    const unsigned short* __restrict__ AloB = Alo + (size_t)m0 * K;
    const unsigned short* __restrict__ BhiB = Bhi + (size_t)n0 * K;
    const unsigned short* __restrict__ BloB = Blo + (size_t)n0 * K;

    v8f acc[2][4];
#pragma unroll
    for (int mf = 0; mf < 2; ++mf)
#pragma unroll
        for (int nf = 0; nf < 4; ++nf)
#pragma unroll
            for (int i = 0; i < 8; ++i) acc[mf][nf][i] = 0.0f;

#if USE_ASYNC_TDM
    // Direct global -> LDS async staging: 12 b128 copies per thread per step
    // (A hi/lo: 2+2, B hi/lo: 4+4); no VGPR data movement.
    auto issue_stage = [&](int k0, int buf) {
#pragma unroll
        for (int i = 0; i < 2; ++i) {
            const int q   = t + 128 * i;        // 0..255
            const int row = q >> 2;             // 4 x 16B per 32-wide row
            const int c8  = (q & 3) << 3;       // ushort col, multiple of 8
            async_cp16(&sAhi[buf][row * LDP + c8], AhiB + (size_t)row * K + k0 + c8);
            async_cp16(&sAlo[buf][row * LDP + c8], AloB + (size_t)row * K + k0 + c8);
        }
#pragma unroll
        for (int i = 0; i < 4; ++i) {
            const int q   = t + 128 * i;        // 0..511
            const int row = q >> 2;
            const int c8  = (q & 3) << 3;
            async_cp16(&sBhi[buf][row * LDP + c8], BhiB + (size_t)row * K + k0 + c8);
            async_cp16(&sBlo[buf][row * LDP + c8], BloB + (size_t)row * K + k0 + c8);
        }
    };
#else
    v4u pah[2], pal[2], pbh[4], pbl[4];
    auto load_stage = [&](int k0) {
#pragma unroll
        for (int i = 0; i < 2; ++i) {
            const int q = t + 128 * i;
            const int row = q >> 2;
            const int c8  = (q & 3) << 3;
            pah[i] = *(const v4u*)(AhiB + (size_t)row * K + k0 + c8);
            pal[i] = *(const v4u*)(AloB + (size_t)row * K + k0 + c8);
        }
#pragma unroll
        for (int i = 0; i < 4; ++i) {
            const int q = t + 128 * i;
            const int row = q >> 2;
            const int c8  = (q & 3) << 3;
            pbh[i] = *(const v4u*)(BhiB + (size_t)row * K + k0 + c8);
            pbl[i] = *(const v4u*)(BloB + (size_t)row * K + k0 + c8);
        }
    };
    auto store_stage = [&](int buf) {
#pragma unroll
        for (int i = 0; i < 2; ++i) {
            const int q = t + 128 * i;
            const int row = q >> 2;
            const int c8  = (q & 3) << 3;
            *(v4u*)(&sAhi[buf][row * LDP + c8]) = pah[i];
            *(v4u*)(&sAlo[buf][row * LDP + c8]) = pal[i];
        }
#pragma unroll
        for (int i = 0; i < 4; ++i) {
            const int q = t + 128 * i;
            const int row = q >> 2;
            const int c8  = (q & 3) << 3;
            *(v4u*)(&sBhi[buf][row * LDP + c8]) = pbh[i];
            *(v4u*)(&sBlo[buf][row * LDP + c8]) = pbl[i];
        }
    };
#endif

    auto compute_stage = [&](int buf) {
        // 16-bit A layout (ISA 7.12.2): lanes 0-15 hold K 0-7 (elems 0-7) and
        // K 16-23 (elems 8-15); lanes 16-31 hold K 8-15 and K 24-31.
        FragBF ahi[2], alo[2];
#pragma unroll
        for (int mf = 0; mf < 2; ++mf) {
            const v4u* pa = (const v4u*)(&sAhi[buf][(wm * 32 + mf * 16 + l16) * LDP + sel * 8]);
            ahi[mf].q[0] = pa[0]; ahi[mf].q[1] = pa[2];       // +16 ushorts = +2 v4u
            const v4u* pb = (const v4u*)(&sAlo[buf][(wm * 32 + mf * 16 + l16) * LDP + sel * 8]);
            alo[mf].q[0] = pb[0]; alo[mf].q[1] = pb[2];
        }
#pragma unroll
        for (int nf = 0; nf < 4; ++nf) {
            FragBF bhi, blo;
            const v4u* pc = (const v4u*)(&sBhi[buf][(wn * 64 + nf * 16 + l16) * LDP + sel * 8]);
            bhi.q[0] = pc[0]; bhi.q[1] = pc[2];
            const v4u* pd = (const v4u*)(&sBlo[buf][(wn * 64 + nf * 16 + l16) * LDP + sel * 8]);
            blo.q[0] = pd[0]; blo.q[1] = pd[2];
#pragma unroll
            for (int mf = 0; mf < 2; ++mf) {
                // bf16x3: hi*hi + hi*lo + lo*hi
                acc[mf][nf] = __builtin_amdgcn_wmma_f32_16x16x32_bf16(
                    false, ahi[mf].v, false, bhi.v, (short)0, acc[mf][nf], false, false);
                acc[mf][nf] = __builtin_amdgcn_wmma_f32_16x16x32_bf16(
                    false, ahi[mf].v, false, blo.v, (short)0, acc[mf][nf], false, false);
                acc[mf][nf] = __builtin_amdgcn_wmma_f32_16x16x32_bf16(
                    false, alo[mf].v, false, bhi.v, (short)0, acc[mf][nf], false, false);
            }
        }
    };

    const int nsteps = K / BK;
#if USE_ASYNC_TDM
    issue_stage(0, 0);
    wait_async0();
    __syncthreads();
    for (int s = 0; s < nsteps; ++s) {
        const int cur = s & 1;
        if (s + 1 < nsteps) issue_stage((s + 1) * BK, cur ^ 1);  // lands during wmma
        compute_stage(cur);
        wait_async0();            // own copies complete (barrier doesn't cover ASYNCcnt)
        __syncthreads();          // everyone's copies visible
    }
#else
    load_stage(0);
    store_stage(0);
    __syncthreads();
    for (int s = 0; s < nsteps; ++s) {
        const int cur = s & 1;
        if (s + 1 < nsteps) load_stage((s + 1) * BK);
        compute_stage(cur);
        if (s + 1 < nsteps) store_stage(cur ^ 1);
        __syncthreads();
    }
#endif

    // epilogue: bias + ReLU; optional f32 and/or bf16-split output.
    // C/D layout: VGPR i -> M = i (+8 for lanes 16-31), N = lane % 16.
#pragma unroll
    for (int mf = 0; mf < 2; ++mf)
#pragma unroll
        for (int nf = 0; nf < 4; ++nf) {
            const int n = n0 + wn * 64 + nf * 16 + l16;
            const float bv = bias[n];
#pragma unroll
            for (int i = 0; i < 8; ++i) {
                const int m = m0 + wm * 32 + mf * 16 + i + sel * 8;
                float v = acc[mf][nf][i] + bv;
                if (relu) v = fmaxf(v, 0.0f);
                const size_t idx = (size_t)m * N + n;
                if (Cf) Cf[idx] = v;
                if (Chi) {
                    unsigned int h, l;
                    bf16_split(v, h, l);
                    Chi[idx] = (unsigned short)h;
                    Clo[idx] = (unsigned short)l;
                }
            }
        }
}

// ---------------------------------------------------------------------------
// Fallback GEMM (small workspace): convert-in-kernel bf16x3, 64x64 tile.
// ---------------------------------------------------------------------------
__global__ __launch_bounds__(128)
void gemm_bf16x3_fb(const float* __restrict__ A,    // [M,K]
                    const float* __restrict__ B,    // [N,K]
                    const float* __restrict__ bias, // [N]
                    float*       __restrict__ C,    // [M,N]
                    int M, int N, int K, int relu)
{
    __shared__ __attribute__((aligned(16))) unsigned short sAhi[BMF * LDP];
    __shared__ __attribute__((aligned(16))) unsigned short sAlo[BMF * LDP];
    __shared__ __attribute__((aligned(16))) unsigned short sBhi[BNF * LDP];
    __shared__ __attribute__((aligned(16))) unsigned short sBlo[BNF * LDP];

    const int t    = threadIdx.x;
    const int wave = t >> 5;
    const int lane = t & 31;
    const int l16  = lane & 15;
    const int sel  = lane >> 4;

    const int m0 = blockIdx.y * BMF;
    const int n0 = blockIdx.x * BNF;

    const float* __restrict__ Ablk = A + (size_t)m0 * K;
    const float* __restrict__ Bblk = B + (size_t)n0 * K;

    v8f acc[4];
#pragma unroll
    for (int nf = 0; nf < 4; ++nf)
#pragma unroll
        for (int i = 0; i < 8; ++i) acc[nf][i] = 0.0f;

    for (int k0 = 0; k0 < K; k0 += BK) {
#pragma unroll
        for (int i = 0; i < 4; ++i) {
            const int q   = t + 128 * i;
            const int row = q >> 3;
            const int c4  = (q & 7) << 2;
            const v4f a4 = *(const v4f*)(Ablk + (size_t)row * K + k0 + c4);
            v2u ph, pl;
            cvt_pack(a4, ph, pl);
            *(v2u*)((unsigned int*)sAhi + row * LDPW + (c4 >> 1)) = ph;
            *(v2u*)((unsigned int*)sAlo + row * LDPW + (c4 >> 1)) = pl;
            const v4f b4 = *(const v4f*)(Bblk + (size_t)row * K + k0 + c4);
            cvt_pack(b4, ph, pl);
            *(v2u*)((unsigned int*)sBhi + row * LDPW + (c4 >> 1)) = ph;
            *(v2u*)((unsigned int*)sBlo + row * LDPW + (c4 >> 1)) = pl;
        }
        __syncthreads();

        FragBF ahi, alo;
        {
            const v4u* pa = (const v4u*)(sAhi + (wave * 16 + l16) * LDP + sel * 8);
            ahi.q[0] = pa[0]; ahi.q[1] = pa[2];
            const v4u* pb = (const v4u*)(sAlo + (wave * 16 + l16) * LDP + sel * 8);
            alo.q[0] = pb[0]; alo.q[1] = pb[2];
        }
#pragma unroll
        for (int nf = 0; nf < 4; ++nf) {
            FragBF bhi, blo;
            const v4u* pc = (const v4u*)(sBhi + (nf * 16 + l16) * LDP + sel * 8);
            bhi.q[0] = pc[0]; bhi.q[1] = pc[2];
            const v4u* pd = (const v4u*)(sBlo + (nf * 16 + l16) * LDP + sel * 8);
            blo.q[0] = pd[0]; blo.q[1] = pd[2];
            acc[nf] = __builtin_amdgcn_wmma_f32_16x16x32_bf16(
                false, ahi.v, false, bhi.v, (short)0, acc[nf], false, false);
            acc[nf] = __builtin_amdgcn_wmma_f32_16x16x32_bf16(
                false, ahi.v, false, blo.v, (short)0, acc[nf], false, false);
            acc[nf] = __builtin_amdgcn_wmma_f32_16x16x32_bf16(
                false, alo.v, false, bhi.v, (short)0, acc[nf], false, false);
        }
        __syncthreads();
    }

#pragma unroll
    for (int nf = 0; nf < 4; ++nf) {
        const int n  = n0 + nf * 16 + l16;
        const float bv = bias[n];
#pragma unroll
        for (int i = 0; i < 8; ++i) {
            const int m = m0 + wave * 16 + i + sel * 8;
            float v = acc[nf][i] + bv;
            if (relu) v = fmaxf(v, 0.0f);
            C[(size_t)m * N + n] = v;
        }
    }
}

// ---------------------------------------------------------------------------
// Stage 4: tiny head GEMMs (N = 84 + 21), f32 VALU.
// ---------------------------------------------------------------------------
__global__ __launch_bounds__(128)
void heads_kernel(const float* __restrict__ h2,     // [R, FC]
                  const float* __restrict__ wloc,   // [84, FC]
                  const float* __restrict__ bloc,
                  const float* __restrict__ wsc,    // [21, FC]
                  const float* __restrict__ bsc,
                  float*       __restrict__ out_loc,  // [R, 84]
                  float*       __restrict__ out_sc)   // [R, 21]
{
    __shared__ __attribute__((aligned(16))) float sh[8][FC];
    const int r0 = blockIdx.x * 8;

    for (int i = threadIdx.x; i < (8 * FC) / 4; i += 128)
        ((v4f*)&sh[0][0])[i] = ((const v4f*)(h2 + (size_t)r0 * FC))[i];
    __syncthreads();

    const int j = threadIdx.x;
    if (j < NLOC + NSCORE) {
        const float* __restrict__ wrow =
            (j < NLOC) ? (wloc + (size_t)j * FC) : (wsc + (size_t)(j - NLOC) * FC);
        float acc[8];
#pragma unroll
        for (int q = 0; q < 8; ++q) acc[q] = 0.0f;

        for (int k = 0; k < FC; k += 4) {
            const v4f w4 = *(const v4f*)(wrow + k);
#pragma unroll
            for (int q = 0; q < 8; ++q)
                acc[q] += w4.x * sh[q][k] + w4.y * sh[q][k + 1] +
                          w4.z * sh[q][k + 2] + w4.w * sh[q][k + 3];
        }
        if (j < NLOC) {
            const float bb = bloc[j];
#pragma unroll
            for (int q = 0; q < 8; ++q)
                out_loc[(size_t)(r0 + q) * NLOC + j] = acc[q] + bb;
        } else {
            const float bb = bsc[j - NLOC];
#pragma unroll
            for (int q = 0; q < 8; ++q)
                out_sc[(size_t)(r0 + q) * NSCORE + (j - NLOC)] = acc[q] + bb;
        }
    }
}

// ---------------------------------------------------------------------------
// Launch. Fast path (~84 MB ws): pool->split -> FC1(wmma,split-out) ->
// FC2(wmma) -> heads. Fallback (~30 MB ws): convert-in-GEMM variant.
// ---------------------------------------------------------------------------
extern "C" void kernel_launch(void* const* d_in, const int* in_sizes, int n_in,
                              void* d_out, int out_size, void* d_ws, size_t ws_size,
                              hipStream_t stream)
{
    (void)in_sizes; (void)n_in; (void)out_size;

    const float* x    = (const float*)d_in[0];
    const float* rois = (const float*)d_in[1];
    const int*   ridx = (const int*)  d_in[2];
    const float* w1   = (const float*)d_in[3];
    const float* b1   = (const float*)d_in[4];
    const float* w2   = (const float*)d_in[5];
    const float* b2   = (const float*)d_in[6];
    const float* wloc = (const float*)d_in[7];
    const float* bloc = (const float*)d_in[8];
    const float* wsc  = (const float*)d_in[9];
    const float* bsc  = (const float*)d_in[10];

    float* outloc = (float*)d_out;                       // [512, 84]
    float* outsc  = outloc + (size_t)RNUM * NLOC;        // [512, 21]

    const size_t sFlat = (size_t)RNUM * INFEAT * 2;      // bytes (one of hi/lo)
    const size_t sW1   = (size_t)FC * INFEAT * 2;
    const size_t sW2   = (size_t)FC * FC * 2;
    const size_t sH1   = (size_t)RNUM * FC * 2;
    const size_t sH2f  = (size_t)RNUM * FC * 4;
    const size_t need  = 2 * sFlat + 2 * sW1 + 2 * sW2 + 2 * sH1 + sH2f;

    if (ws_size >= need) {
        // ---------------- fast path: everything pre-split to bf16 hi/lo
        char* p = (char*)d_ws;
        unsigned short* flatHi = (unsigned short*)p;            p += sFlat;
        unsigned short* flatLo = (unsigned short*)p;            p += sFlat;
        unsigned short* w1hi   = (unsigned short*)p;            p += sW1;
        unsigned short* w1lo   = (unsigned short*)p;            p += sW1;
        unsigned short* w2hi   = (unsigned short*)p;            p += sW2;
        unsigned short* w2lo   = (unsigned short*)p;            p += sW2;
        unsigned short* h1hi   = (unsigned short*)p;            p += sH1;
        unsigned short* h1lo   = (unsigned short*)p;            p += sH1;
        float*          h2     = (float*)p;

        roi_pool_kernel<<<RNUM, 256, 0, stream>>>(x, rois, ridx,
                                                  nullptr, flatHi, flatLo);

        const int n4w1 = (FC * INFEAT) / 4;
        split_kernel<<<(n4w1 + 255) / 256, 256, 0, stream>>>(w1, w1hi, w1lo, n4w1);
        const int n4w2 = (FC * FC) / 4;
        split_kernel<<<(n4w2 + 255) / 256, 256, 0, stream>>>(w2, w2hi, w2lo, n4w2);

        dim3 g(FC / BN, RNUM / BM);   // (8, 8)
        gemm_bf16x3_pre<<<g, 128, 0, stream>>>(flatHi, flatLo, w1hi, w1lo, b1,
                                               nullptr, h1hi, h1lo,
                                               RNUM, FC, INFEAT, 1);
        gemm_bf16x3_pre<<<g, 128, 0, stream>>>(h1hi, h1lo, w2hi, w2lo, b2,
                                               h2, nullptr, nullptr,
                                               RNUM, FC, FC, 1);

        heads_kernel<<<RNUM / 8, 128, 0, stream>>>(h2, wloc, bloc, wsc, bsc,
                                                   outloc, outsc);
    } else {
        // ---------------- fallback: f32 staging, convert inside GEMM
        float* flat = (float*)d_ws;                          // [512, 12544]
        float* h1   = flat + (size_t)RNUM * INFEAT;          // [512, 1024]
        float* h2   = h1   + (size_t)RNUM * FC;              // [512, 1024]

        roi_pool_kernel<<<RNUM, 256, 0, stream>>>(x, rois, ridx,
                                                  flat, nullptr, nullptr);

        dim3 g1(FC / BNF, RNUM / BMF);   // (16, 8)
        gemm_bf16x3_fb<<<g1, 128, 0, stream>>>(flat, w1, b1, h1, RNUM, FC, INFEAT, 1);
        gemm_bf16x3_fb<<<g1, 128, 0, stream>>>(h1,   w2, b2, h2, RNUM, FC, FC,     1);

        heads_kernel<<<RNUM / 8, 128, 0, stream>>>(h2, wloc, bloc, wsc, bsc,
                                                   outloc, outsc);
    }
}
